// MambaBlock_18777597018877
// MI455X (gfx1250) — matrix operand order repaired and checked by reading, not compiled, not run on hardware
//
#include <hip/hip_runtime.h>
#include <hip/hip_bf16.h>

// ---------------------------------------------------------------------------
// MambaBlock (LayerNorm -> diagonal S4D scan -> residual) on CDNA5 / gfx1250.
//
// Strategy: chunked (SSD-style) reformulation of the diagonal scan so all the
// heavy math runs on v_wmma_f32_16x16x32_f16 with fp32 accumulation:
//   per channel d, superchunk of Q=32 steps, batch b=0..15 = WMMA N dim:
//     Y(16x16 tile)  = G * U        (Toeplitz of k[tau] = sum_n C a^tau Bbar)
//                    + E * h0       (E[t,n] = C_n a_n^{t+1}, K=64 -> 2 WMMAs)
//     h <- a^32 (.) h + W * U       (W[n,s] = a_n^{31-s} Bbar_n, K=32 full)
// L is split into 16 segments (blocked scan): phaseA computes per-segment
// local states (h_in=0), phaseB combines them sequentially (a^256 decay),
// phaseC replays the scan with correct incoming state and writes the fused
// output  out = x + Dskip*u + y.
// ---------------------------------------------------------------------------

typedef __attribute__((ext_vector_type(16))) _Float16 v16h;
typedef __attribute__((ext_vector_type(8)))  _Float16 v8h;
typedef __attribute__((ext_vector_type(8)))  float    v8f;

#define WMMA_F16(A, Bm, C) \
  __builtin_amdgcn_wmma_f32_16x16x32_f16(false, (A), false, (Bm), (short)0, (C), false, false)

namespace {
constexpr int Bn   = 16;    // batch
constexpr int Ln   = 4096;  // sequence length
constexpr int Dm   = 256;   // channels
constexpr int Ns   = 64;    // states per channel
constexpr int NSEG = 16;    // segments for blocked scan
constexpr int SEGL = Ln / NSEG;   // 256
constexpr int NSC  = SEGL / 32;   // 8 superchunks per segment

// per-channel fragment table: 10 frags x 512 halves (A-fragment lane layout)
constexpr int FRAG_HALVES = 512;
constexpr int NFRAG       = 10;            // G1,G2,E1,E2,E3,E4,W0..W3
constexpr int CH_HALVES   = NFRAG * FRAG_HALVES;  // 5120

// workspace layout (bytes)
constexpr size_t OFF_U16   = 0;                                   // B*L*D f16
constexpr size_t SZ_U16    = (size_t)Bn * Ln * Dm * 2;            // 33.55 MB
constexpr size_t OFF_FRAGS = OFF_U16 + SZ_U16;
constexpr size_t SZ_FRAGS  = (size_t)Dm * CH_HALVES * 2;          // 2.62 MB
constexpr size_t OFF_A32   = OFF_FRAGS + SZ_FRAGS;                // [D][128] f32 (a^32 | a^256)
constexpr size_t SZ_A32    = (size_t)Dm * 128 * 4;
constexpr size_t OFF_HLOC  = OFF_A32 + SZ_A32;                    // [seg][d][b][n] f32
constexpr size_t SZ_H      = (size_t)NSEG * Dm * Bn * Ns * 4;     // 16.78 MB
constexpr size_t OFF_HIN   = OFF_HLOC + SZ_H;                     // [seg][d][b][n] f32
} // namespace

// ---------------------------------------------------------------------------
// Kernel 0: per-channel parameter setup.  Builds k[tau], then writes the ten
// 16x32 f16 A-matrices in this file's A-fragment order (contiguous k, as the B operands use):
//   element (lane, i):  M = lane%16,  K = i + 16*(lane/16)
// ---------------------------------------------------------------------------
__global__ void k_setup(const float* __restrict__ A_log,
                        const float* __restrict__ log_dt,
                        const float* __restrict__ Bp,
                        const float* __restrict__ Cp,
                        _Float16* __restrict__ frags,
                        float* __restrict__ a32g) {
  const int d = blockIdx.x;
  const int tid = threadIdx.x;
  __shared__ float sa[Ns], sB[Ns], sC[Ns], sk[32];

  if (tid < Ns) {
    float dt = expf(log_dt[d]);
    float a  = expf(-expf(A_log[d * Ns + tid]) * dt);
    sa[tid] = a;
    sB[tid] = Bp[d * Ns + tid] * dt;
    sC[tid] = Cp[d * Ns + tid];
    float a32 = powf(a, 32.0f);
    a32g[d * 128 + tid]      = a32;            // a^32
    a32g[d * 128 + 64 + tid] = powf(a32, 8.0f);// a^256
  }
  __syncthreads();
  if (tid < 32) {
    float s = 0.0f;
    for (int n = 0; n < Ns; ++n) s += sC[n] * powf(sa[n], (float)tid) * sB[n];
    sk[tid] = s;
  }
  __syncthreads();

  for (int e = tid; e < CH_HALVES; e += blockDim.x) {
    const int f    = e >> 9;          // fragment id
    const int pos  = e & 511;
    const int lane = pos >> 4;
    const int i    = pos & 15;
    const int r    = lane & 15;
    const int half = lane >> 4;
    const int K    = i + 16 * half;
    float v = 0.0f;
    if (f == 0) {                      // G1: first 16 outputs, s<16 causal
      if (K < 16 && K <= r) v = sk[r - K];
    } else if (f == 1) {               // G2: outputs t=16+r over full 32 inputs
      if (K < 16) v = sk[16 + r - K];
      else { int sp = K - 16; if (sp <= r) v = sk[r - sp]; }
    } else if (f <= 5) {               // E tiles: C_n * a_n^{t+1}
      int t  = r + ((f >= 4) ? 16 : 0);
      int n  = ((f - 2) & 1) * 32 + K;
      v = sC[n] * powf(sa[n], (float)(t + 1));
    } else {                           // W tiles: a_n^{31-s} * Bbar_n
      int tl = f - 6;
      int n  = tl * 16 + r;
      v = powf(sa[n], (float)(31 - K)) * sB[n];
    }
    frags[(size_t)d * CH_HALVES + e] = (_Float16)v;
  }
}

// ---------------------------------------------------------------------------
// Kernel 1: LayerNorm, one wave (32 lanes) per (b,t) row, coalesced; emits u
// in f16 (WMMA operand precision, halves scan-pass read traffic).
// ---------------------------------------------------------------------------
__global__ void k_layernorm(const float* __restrict__ x,
                            const float* __restrict__ w,
                            const float* __restrict__ bias,
                            _Float16* __restrict__ u16) {
  const int row  = blockIdx.x * 8 + (threadIdx.x >> 5);
  const int lane = threadIdx.x & 31;
  const float* xr = x + (size_t)row * Dm;
  float v[8], s = 0.0f, sq = 0.0f;
#pragma unroll
  for (int k = 0; k < 8; ++k) {
    v[k] = xr[lane + 32 * k];
    s += v[k]; sq += v[k] * v[k];
  }
#pragma unroll
  for (int off = 16; off > 0; off >>= 1) {
    s  += __shfl_xor(s, off);
    sq += __shfl_xor(sq, off);
  }
  const float mu  = s * (1.0f / Dm);
  const float var = sq * (1.0f / Dm) - mu * mu;
  const float inv = rsqrtf(var + 1e-5f);
#pragma unroll
  for (int k = 0; k < 8; ++k) {
    int c = lane + 32 * k;
    float u = (v[k] - mu) * inv * w[c] + bias[c];
    u16[(size_t)row * Dm + c] = (_Float16)u;
  }
}

// ---------------------------------------------------------------------------
// Kernel 2 (phase A): per (segment, channel) wave computes the segment-local
// final state with h_in = 0.  Only the state-advance WMMAs (4 per superchunk).
// ---------------------------------------------------------------------------
__global__ void k_stateA(const _Float16* __restrict__ u16,
                         const _Float16* __restrict__ frags,
                         const float* __restrict__ a32g,
                         float* __restrict__ h_loc) {
  const int w    = blockIdx.x * 8 + (threadIdx.x >> 5);
  const int d    = w & (Dm - 1);
  const int seg  = w >> 8;
  const int lane = threadIdx.x & 31;
  const int r    = lane & 15;
  const int half = lane >> 4;

  const v16h* Fp = (const v16h*)(frags + (size_t)d * CH_HALVES);
  v16h Wfr[4];
#pragma unroll
  for (int tl = 0; tl < 4; ++tl) Wfr[tl] = Fp[(6 + tl) * 32 + lane];
  const float* ap = a32g + d * 128;
  v8f a32t[4];
#pragma unroll
  for (int tl = 0; tl < 4; ++tl) a32t[tl] = *(const v8f*)(ap + tl * 16 + 8 * half);

  v8f h[4];
#pragma unroll
  for (int tl = 0; tl < 4; ++tl) h[tl] = 0;

  const int tbase = seg * SEGL;
  for (int sc = 0; sc < NSC; ++sc) {
    const int t0 = tbase + sc * 32;
    v16h U;
    const _Float16* up = u16 + (((size_t)r * Ln + t0 + half * 16) * Dm + d);
#pragma unroll
    for (int i = 0; i < 16; ++i) U[i] = up[(size_t)i * Dm];
#pragma unroll
    for (int tl = 0; tl < 4; ++tl) {
      v8f c;
#pragma unroll
      for (int j = 0; j < 8; ++j) c[j] = a32t[tl][j] * h[tl][j];
      h[tl] = WMMA_F16(Wfr[tl], U, c);
    }
  }
  float* hp = h_loc + (((size_t)(seg * Dm + d) * Bn) + r) * Ns;
#pragma unroll
  for (int tl = 0; tl < 4; ++tl) *(v8f*)(hp + tl * 16 + 8 * half) = h[tl];
}

// ---------------------------------------------------------------------------
// Kernel 3 (phase B): sequential combine of segment states per channel:
//   h_in[s] = a^256 (.) h_in[s-1] + h_loc[s-1],  h_in[0] = 0
// ---------------------------------------------------------------------------
__global__ void k_combine(const float* __restrict__ h_loc,
                          const float* __restrict__ a32g,
                          float* __restrict__ h_in) {
  const int d = blockIdx.x;
  for (int e = threadIdx.x; e < Bn * Ns; e += blockDim.x) {
    const int n = e & (Ns - 1);
    const float a256 = a32g[d * 128 + 64 + n];
    float val = 0.0f;
    for (int s = 0; s < NSEG; ++s) {
      const size_t idx = (size_t)(s * Dm + d) * (Bn * Ns) + e;
      h_in[idx] = val;
      val = a256 * val + h_loc[idx];
    }
  }
}

// ---------------------------------------------------------------------------
// Kernel 4 (phase C): full corrected scan + fused output.
// 10 WMMAs per superchunk per wave; h -> f16 B-fragment via per-wave LDS.
// ---------------------------------------------------------------------------
__global__ void __launch_bounds__(256)
k_scanC(const float* __restrict__ x,
        const _Float16* __restrict__ u16,
        const _Float16* __restrict__ frags,
        const float* __restrict__ a32g,
        const float* __restrict__ h_in,
        const float* __restrict__ Dskip,
        float* __restrict__ out) {
  __shared__ _Float16 hstage[8 * Bn * Ns];   // 8 waves x 2KB, per-wave private
  const int wib  = threadIdx.x >> 5;
  const int w    = blockIdx.x * 8 + wib;
  const int d    = w & (Dm - 1);
  const int seg  = w >> 8;
  const int lane = threadIdx.x & 31;
  const int r    = lane & 15;
  const int half = lane >> 4;

  const v16h* Fp = (const v16h*)(frags + (size_t)d * CH_HALVES);
  const v16h G1 = Fp[0 * 32 + lane], G2 = Fp[1 * 32 + lane];
  const v16h E1 = Fp[2 * 32 + lane], E2 = Fp[3 * 32 + lane];
  const v16h E3 = Fp[4 * 32 + lane], E4 = Fp[5 * 32 + lane];
  v16h Wfr[4];
#pragma unroll
  for (int tl = 0; tl < 4; ++tl) Wfr[tl] = Fp[(6 + tl) * 32 + lane];

  const float* ap = a32g + d * 128;
  v8f a32t[4];
#pragma unroll
  for (int tl = 0; tl < 4; ++tl) a32t[tl] = *(const v8f*)(ap + tl * 16 + 8 * half);
  const float Dk = Dskip[d];

  v8f h[4];
  const float* hp = h_in + (((size_t)(seg * Dm + d) * Bn) + r) * Ns;
#pragma unroll
  for (int tl = 0; tl < 4; ++tl) h[tl] = *(const v8f*)(hp + tl * 16 + 8 * half);

  _Float16* ls = hstage + wib * (Bn * Ns);   // [b][n] layout
  const int tbase = seg * SEGL;

  for (int sc = 0; sc < NSC; ++sc) {
    const int t0 = tbase + sc * 32;

    // U B-fragment: (K = s = 16*half + i, N = b = r)
    v16h U;
    const _Float16* up = u16 + (((size_t)r * Ln + t0 + half * 16) * Dm + d);
#pragma unroll
    for (int i = 0; i < 16; ++i) U[i] = up[(size_t)i * Dm];

    // stage h (f32 D-tile layout) to LDS as f16 [b][n]
#pragma unroll
    for (int tl = 0; tl < 4; ++tl) {
      v8h hv;
#pragma unroll
      for (int j = 0; j < 8; ++j) hv[j] = (_Float16)h[tl][j];
      *(v8h*)(ls + r * Ns + tl * 16 + 8 * half) = hv;
    }
    asm volatile("s_wait_dscnt 0" ::: "memory");  // same-wave LDS ordering
    // re-read as B-fragments (K = n = 16*half + i [+32], N = b = r)
    const v16h hB0 = *(const v16h*)(ls + r * Ns + half * 16);
    const v16h hB1 = *(const v16h*)(ls + r * Ns + 32 + half * 16);
    asm volatile("" ::: "memory");

    v8f Y1 = 0, Y2 = 0;
    Y1 = WMMA_F16(G1, U, Y1);
    Y1 = WMMA_F16(E1, hB0, Y1);
    Y1 = WMMA_F16(E2, hB1, Y1);
    Y2 = WMMA_F16(G2, U, Y2);
    Y2 = WMMA_F16(E3, hB0, Y2);
    Y2 = WMMA_F16(E4, hB1, Y2);

    // state advance: h <- a^32 (.) h + W * U   (fp32 carried state)
#pragma unroll
    for (int tl = 0; tl < 4; ++tl) {
      v8f c;
#pragma unroll
      for (int j = 0; j < 8; ++j) c[j] = a32t[tl][j] * h[tl][j];
      h[tl] = WMMA_F16(Wfr[tl], U, c);
    }

    // fused output: out = x + Dskip*u + y   (D-tile: M = t-offset, N = b)
#pragma unroll
    for (int i = 0; i < 8; ++i) {
      const int t = t0 + i + 8 * half;
      const size_t g  = ((size_t)r * Ln + t) * Dm + d;
      const size_t g2 = g + 16 * Dm;
      out[g]  = x[g]  + Dk * (float)u16[g]  + Y1[i];
      out[g2] = x[g2] + Dk * (float)u16[g2] + Y2[i];
    }
  }
}

// ---------------------------------------------------------------------------
extern "C" void kernel_launch(void* const* d_in, const int* in_sizes, int n_in,
                              void* d_out, int out_size, void* d_ws, size_t ws_size,
                              hipStream_t stream) {
  (void)in_sizes; (void)n_in; (void)out_size; (void)ws_size;
  const float* x      = (const float*)d_in[0];
  const float* norm_w = (const float*)d_in[1];
  const float* norm_b = (const float*)d_in[2];
  const float* A_log  = (const float*)d_in[3];
  const float* log_dt = (const float*)d_in[4];
  const float* Bp     = (const float*)d_in[5];
  const float* Cp     = (const float*)d_in[6];
  const float* Dskip  = (const float*)d_in[7];
  float* out = (float*)d_out;

  char* ws = (char*)d_ws;
  _Float16* u16   = (_Float16*)(ws + OFF_U16);
  _Float16* frg   = (_Float16*)(ws + OFF_FRAGS);
  float*    a32g  = (float*)(ws + OFF_A32);
  float*    h_loc = (float*)(ws + OFF_HLOC);
  float*    h_in  = (float*)(ws + OFF_HIN);

  k_setup    <<<Dm, 256, 0, stream>>>(A_log, log_dt, Bp, Cp, frg, a32g);
  k_layernorm<<<(Bn * Ln) / 8, 256, 0, stream>>>(x, norm_w, norm_b, u16);
  k_stateA   <<<(NSEG * Dm) / 8, 256, 0, stream>>>(u16, frg, a32g, h_loc);
  k_combine  <<<Dm, 256, 0, stream>>>(h_loc, a32g, h_in);
  k_scanC    <<<(NSEG * Dm) / 8, 256, 0, stream>>>(x, u16, frg, a32g, h_in, Dskip, out);
}
